// IAI_83897891160148
// MI455X (gfx1250) — matrix operand order repaired and checked by reading, not compiled, not run on hardware
//
#include <hip/hip_runtime.h>

// ---------------- CDNA5 WMMA types / helpers ----------------
typedef __attribute__((ext_vector_type(16))) _Float16 v16h;
typedef __attribute__((ext_vector_type(8)))  _Float16 v8h;
typedef __attribute__((ext_vector_type(8)))  float    v8f;

#define WMMA_F16(a, b, c) \
  __builtin_amdgcn_wmma_f32_16x16x32_f16(false, (a), false, (b), (short)0, (c), false, false)

static constexpr int CH  = 256;   // channels
static constexpr int HWN = 4096;  // tokens (64*64)
static constexpr int NH  = 8;     // heads
static constexpr int DK  = 32;    // head dim

__device__ __forceinline__ v16h cat16(v8h lo, v8h hi) {
  v16h r;
#pragma unroll
  for (int i = 0; i < 8; ++i) { r[i] = lo[i]; r[i + 8] = hi[i]; }
  return r;
}

// A-matrix fragment (16x32 f16, MxK): k(i) = 16*(i>>3) + 8*half + (i&7)
__device__ __forceinline__ v16h load_a_frag(const _Float16* row, int half) {
  v8h lo = *(const v8h*)(row + 8 * half);
  v8h hi = *(const v8h*)(row + 16 + 8 * half);
  return cat16(lo, hi);
}

// B-matrix fragment (32x16 f16, KxN): k(i) = 16*(i>>3) + 8*half + (i&7)
__device__ __forceinline__ v16h load_b_frag(const _Float16* col, int half) {
  v8h lo = *(const v8h*)(col + 8 * half);
  v8h hi = *(const v8h*)(col + 16 + 8 * half);
  return cat16(lo, hi);
}

// ---- CDNA5 async global->LDS copy (ASYNCcnt path), per-lane 16B ----
// GV mode: vdst = LDS byte address (low 32 bits of flat shared ptr), vaddr = 64-bit global.
__device__ __forceinline__ void async_b128(const void* g, void* l) {
  asm volatile("global_load_async_to_lds_b128 %0, %1, off"
               :: "v"((unsigned int)(uintptr_t)l), "v"((unsigned long long)(uintptr_t)g)
               : "memory");
}
__device__ __forceinline__ void wait_async0() {
  asm volatile("s_wait_asynccnt 0x0" ::: "memory");
}

// ---------------- Kernel 1: QKV projection (per-pixel dense) ----------------
// out[o,p] = sum_c W[o,c]*in[c,p] + b[o].
// vtrans==0: f16 head-major [b,h,n,dk] (Q pre-scaled by dk^-1/2, and K).
// vtrans==1: f16 d-major    [b,h,dk,n] (V, so flash can async-copy V tiles).
__global__ void __launch_bounds__(128)
iai_proj_qkv(const float* __restrict__ in, const float* __restrict__ W,
             const float* __restrict__ bias, _Float16* __restrict__ outw,
             float oscale, int vtrans) {
  __shared__ _Float16 in_t[64][40];  // [pixel][k]  (80B rows, 16B aligned)
  __shared__ _Float16 w_s[64][40];   // [m][k]
  const int b        = blockIdx.z;
  const int pix_base = blockIdx.x * 64;
  const int ch_base  = blockIdx.y * 64;
  const int tid  = threadIdx.x;
  const int wave = tid >> 5;
  const int lane = tid & 31;
  const int half = lane >> 4;
  const int l15  = lane & 15;

  const float* inb = in + (size_t)b * CH * HWN;

  v8f acc[4] = {};
  for (int kc = 0; kc < CH; kc += 32) {
    __syncthreads();
    // input tile 32(k) x 64(n): per-thread gather of 16 strided f32 (issues as a
    // load clause), convert, two 16B LDS stores into transposed [n][k].
    {
      int n = tid & 63, kh = (tid >> 6) * 16;
      const float* src = inb + (size_t)(kc + kh) * HWN + pix_base + n;
      float tf[16];
#pragma unroll
      for (int i = 0; i < 16; ++i) tf[i] = src[(size_t)i * HWN];
      v8h lo, hi;
#pragma unroll
      for (int i = 0; i < 8; ++i) { lo[i] = (_Float16)tf[i]; hi[i] = (_Float16)tf[i + 8]; }
      *(v8h*)&in_t[n][kh]     = lo;
      *(v8h*)&in_t[n][kh + 8] = hi;
    }
    // weight tile 64(m) x 32(k): 16 contiguous f32 per thread (b128 loads), 2 LDS stores.
    {
      int m = tid >> 1, part = tid & 1;
      const float* wsrc = W + (size_t)(ch_base + m) * CH + kc + part * 16;
      float tw[16];
#pragma unroll
      for (int i = 0; i < 16; ++i) tw[i] = wsrc[i];
      v8h lo, hi;
#pragma unroll
      for (int i = 0; i < 8; ++i) { lo[i] = (_Float16)tw[i]; hi[i] = (_Float16)tw[i + 8]; }
      *(v8h*)&w_s[m][part * 16]     = lo;
      *(v8h*)&w_s[m][part * 16 + 8] = hi;
    }
    __syncthreads();
    v16h a = load_a_frag(&w_s[16 * wave + l15][0], half);
#pragma unroll
    for (int nt = 0; nt < 4; ++nt) {
      v16h bf = load_b_frag(&in_t[16 * nt + l15][0], half);
      acc[nt] = WMMA_F16(a, bf, acc[nt]);
    }
  }
  // epilogue
  const int m_base = ch_base + 16 * wave + 8 * half;  // multiple of 8 -> one head
  const int head = m_base >> 5, d0 = m_base & 31;
  if (vtrans) {
    // d-major: out[(bh*DK + d)*HWN + pix]
#pragma unroll
    for (int nt = 0; nt < 4; ++nt) {
      int pix = pix_base + 16 * nt + l15;
#pragma unroll
      for (int j = 0; j < 8; ++j) {
        float v = (acc[nt][j] + bias[m_base + j]) * oscale;
        outw[((size_t)(b * NH + head) * DK + d0 + j) * HWN + pix] = (_Float16)v;
      }
    }
  } else {
    // head-major: 8 contiguous d per lane = one 16B store
#pragma unroll
    for (int nt = 0; nt < 4; ++nt) {
      int pix = pix_base + 16 * nt + l15;
      v8h pack;
#pragma unroll
      for (int j = 0; j < 8; ++j)
        pack[j] = (_Float16)((acc[nt][j] + bias[m_base + j]) * oscale);
      size_t row = (size_t)(b * NH + head) * HWN + pix;
      *(v8h*)(outw + row * DK + d0) = pack;
    }
  }
}

// ---------------- Kernel 2: flash attention ----------------
// 256 threads = 8 waves; each wave owns 16 query rows; 64-key tiles staged in LDS
// via GLOBAL_LOAD_ASYNC_TO_LDS_B128 (K head-major, V d-major -> no transpose needed).
__global__ void __launch_bounds__(256)
iai_flash_attn(const _Float16* __restrict__ Q, const _Float16* __restrict__ K,
               const _Float16* __restrict__ Vt, _Float16* __restrict__ O) {
  __shared__ _Float16 k_t[64][40];      // [key][d]   S B-fragments
  __shared__ _Float16 v_t[32][72];      // [d][key]   PV B-fragments (144B rows)
  __shared__ _Float16 p_t[8][16][72];   // wave-private P [m][key]
  const int bh   = blockIdx.y;
  const int tid  = threadIdx.x;
  const int wave = tid >> 5;
  const int lane = tid & 31;
  const int half = lane >> 4;
  const int l15  = lane & 15;

  const _Float16* Qb  = Q  + (size_t)bh * HWN * DK;
  const _Float16* Kb  = K  + (size_t)bh * HWN * DK;
  const _Float16* Vbt = Vt + (size_t)bh * DK * HWN;   // [d][n]

  const int qrow = blockIdx.x * 128 + 16 * wave;
  const v16h qf = load_a_frag(Qb + (size_t)(qrow + l15) * DK, half);  // resident

  float mrow[8], lrow[8];
  v8f oacc[2] = {};
#pragma unroll
  for (int j = 0; j < 8; ++j) { mrow[j] = -1e30f; lrow[j] = 0.f; }

  for (int kt = 0; kt < HWN; kt += 64) {
    __syncthreads();
    // async-stage K (64x32) and V^T (32x64): one 16B copy each per lane
    {
      int key = tid >> 2, kp = tid & 3;                       // 256 x 16B = 4KB
      async_b128(Kb + (size_t)(kt + key) * DK + kp * 8, &k_t[key][kp * 8]);
      int d = tid >> 3, vp = tid & 7;                         // 256 x 16B = 4KB
      async_b128(Vbt + (size_t)d * HWN + kt + vp * 8, &v_t[d][vp * 8]);
    }
    if (kt + 64 < HWN) {  // warm next tile toward GL2 (global_prefetch_b8)
      __builtin_prefetch(Kb + (size_t)(kt + 64) * DK + (tid & 63) * 16, 0, 0);
      __builtin_prefetch(Vbt + (size_t)(tid & 31) * HWN + kt + 64, 0, 0);
    }
    wait_async0();
    __syncthreads();

    // S = Q*K^T (Q carries dk^-1/2): 4 WMMAs over 64 keys
    v8f s[4];
#pragma unroll
    for (int nt = 0; nt < 4; ++nt) {
      v16h bf = load_b_frag(&k_t[16 * nt + l15][0], half);
      v8f z = {};
      s[nt] = WMMA_F16(qf, bf, z);
    }

    // online softmax; row m=j+8*half lives across this half's 16 lanes
#pragma unroll
    for (int j = 0; j < 8; ++j) {
      float t = fmaxf(fmaxf(s[0][j], s[1][j]), fmaxf(s[2][j], s[3][j]));
#pragma unroll
      for (int off = 8; off >= 1; off >>= 1) t = fmaxf(t, __shfl_xor(t, off, 32));
      float mn   = fmaxf(mrow[j], t);
      float corr = __expf(mrow[j] - mn);
      mrow[j] = mn;
      float rs = 0.f;
#pragma unroll
      for (int nt = 0; nt < 4; ++nt) {
        float p = __expf(s[nt][j] - mn);
        s[nt][j] = p;
        rs += p;
      }
#pragma unroll
      for (int off = 8; off >= 1; off >>= 1) rs += __shfl_xor(rs, off, 32);
      lrow[j] = lrow[j] * corr + rs;
      oacc[0][j] *= corr;
      oacc[1][j] *= corr;
    }

    // P: C-layout -> A-layout via wave-private LDS round trip
#pragma unroll
    for (int nt = 0; nt < 4; ++nt) {
      int key = 16 * nt + l15;
#pragma unroll
      for (int j = 0; j < 8; ++j)
        p_t[wave][j + 8 * half][key] = (_Float16)s[nt][j];
    }
    // O += P*V : 2 key-chunks x 2 d-halves = 4 WMMAs
#pragma unroll
    for (int c2 = 0; c2 < 2; ++c2) {
      v16h pa = load_a_frag(&p_t[wave][l15][32 * c2], half);
#pragma unroll
      for (int dt = 0; dt < 2; ++dt) {
        v16h vb = load_b_frag(&v_t[16 * dt + l15][32 * c2], half);
        oacc[dt] = WMMA_F16(pa, vb, oacc[dt]);
      }
    }
  }

  // normalize, store f16 head-major
#pragma unroll
  for (int dt = 0; dt < 2; ++dt) {
    int d = 16 * dt + l15;
#pragma unroll
    for (int j = 0; j < 8; ++j) {
      int m = j + 8 * half;
      O[((size_t)bh * HWN + qrow + m) * DK + d] = (_Float16)(oacc[dt][j] / lrow[j]);
    }
  }
}

// ---------------- Kernel 3: output projection ----------------
__global__ void __launch_bounds__(128)
iai_proj_out(const _Float16* __restrict__ Ow, const float* __restrict__ W,
             const float* __restrict__ bias, float* __restrict__ out) {
  __shared__ _Float16 in_t[64][40];
  __shared__ _Float16 w_s[64][40];
  const int b        = blockIdx.z;
  const int pix_base = blockIdx.x * 64;
  const int ch_base  = blockIdx.y * 64;
  const int tid  = threadIdx.x;
  const int wave = tid >> 5;
  const int lane = tid & 31;
  const int half = lane >> 4;
  const int l15  = lane & 15;

  v8f acc[4] = {};
  for (int kc = 0; kc < CH; kc += 32) {
    __syncthreads();
    // attention-output tile (pure f16 copy): async global->LDS, 2 x 16B per thread
    const _Float16* src = Ow + (size_t)(b * NH + (kc >> 5)) * HWN * DK;
#pragma unroll
    for (int r = 0; r < 2; ++r) {
      int c = tid + r * 128;                 // 0..255
      int n = c >> 2, part = c & 3;
      async_b128(src + (size_t)(pix_base + n) * DK + part * 8, &in_t[n][part * 8]);
    }
    // weight tile: contiguous f32 -> f16
    {
      int m = tid >> 1, part = tid & 1;
      const float* wsrc = W + (size_t)(ch_base + m) * CH + kc + part * 16;
      float tw[16];
#pragma unroll
      for (int i = 0; i < 16; ++i) tw[i] = wsrc[i];
      v8h lo, hi;
#pragma unroll
      for (int i = 0; i < 8; ++i) { lo[i] = (_Float16)tw[i]; hi[i] = (_Float16)tw[i + 8]; }
      *(v8h*)&w_s[m][part * 16]     = lo;
      *(v8h*)&w_s[m][part * 16 + 8] = hi;
    }
    wait_async0();
    __syncthreads();
    v16h a = load_a_frag(&w_s[16 * wave + l15][0], half);
#pragma unroll
    for (int nt = 0; nt < 4; ++nt) {
      v16h bf = load_b_frag(&in_t[16 * nt + l15][0], half);
      acc[nt] = WMMA_F16(a, bf, acc[nt]);
    }
  }
#pragma unroll
  for (int nt = 0; nt < 4; ++nt) {
    int pix = pix_base + 16 * nt + l15;
#pragma unroll
    for (int j = 0; j < 8; ++j) {
      int o = ch_base + 16 * wave + 8 * half + j;
      out[((size_t)b * CH + o) * HWN + pix] = acc[nt][j] + bias[o];
    }
  }
}

// ---------------- Host launcher ----------------
extern "C" void kernel_launch(void* const* d_in, const int* in_sizes, int n_in,
                              void* d_out, int out_size, void* d_ws, size_t ws_size,
                              hipStream_t stream) {
  const float* x    = (const float*)d_in[0];
  const float* f_lu = (const float*)d_in[1];
  const float* Wq   = (const float*)d_in[2];
  const float* bq   = (const float*)d_in[3];
  const float* Wk   = (const float*)d_in[4];
  const float* bk   = (const float*)d_in[5];
  const float* Wv   = (const float*)d_in[6];
  const float* bv   = (const float*)d_in[7];
  const float* Wo   = (const float*)d_in[8];
  const float* bo   = (const float*)d_in[9];
  float* out = (float*)d_out;

  const size_t per = (size_t)2 * NH * HWN * DK;  // 2M f16 elems per tensor
  _Float16* Qw = (_Float16*)d_ws;                // [bh][n][dk]
  _Float16* Kw = Qw + per;                       // [bh][n][dk]
  _Float16* Vw = Kw + per;                       // [bh][dk][n]  (transposed for async V)
  _Float16* Ow = Vw + per;                       // [bh][n][dk]

  dim3 gp(HWN / 64, CH / 64, 2);
  const float scale = 0.17677669529663687f;      // dk^-0.5, folded into Q
  iai_proj_qkv<<<gp, 128, 0, stream>>>(f_lu, Wq, bq, Qw, scale, 0);
  iai_proj_qkv<<<gp, 128, 0, stream>>>(x,    Wk, bk, Kw, 1.0f,  0);
  iai_proj_qkv<<<gp, 128, 0, stream>>>(x,    Wv, bv, Vw, 1.0f,  1);

  dim3 gf(HWN / 128, 2 * NH, 1);
  iai_flash_attn<<<gf, 256, 0, stream>>>(Qw, Kw, Vw, Ow);

  iai_proj_out<<<gp, 128, 0, stream>>>(Ow, Wo, bo, out);
}
